// SimpleRNNModel_5506148074055
// MI455X (gfx1250) — matrix operand order repaired and checked by reading, not compiled, not run on hardware
//
#include <hip/hip_runtime.h>
#include <stdint.h>

// ---------------------------------------------------------------------------
// 2-layer GRU (B=32,T=2048,IN=512,H=256) + FC(41) for MI455X / gfx1250.
//   Phase 1: gx0 = f16(x) @ f16(W_ih0)^T + b_ih0      (WMMA GEMM, f16 out)
//   Phase 2: persistent scan: 2 batch groups x 4 column-WGPs,
//            W_hh slice (96KB) LDS-resident, ping-pong f16 h state,
//            4-WG sync chain (atomic + hw cluster barrier)
//   Phase 3: gx1 = h1 @ W_ih1^T + b_ih1               (WMMA GEMM, f16 out)
//   Phase 4: scan layer 1
//   Phase 5: logits = h2 @ W_fc^T + b_fc              (WMMA GEMM, f32 out)
// ---------------------------------------------------------------------------

#define GB    32         // batch
#define GT    2048       // time steps
#define GIN   512
#define GH    256
#define G3    768        // 3*H
#define GNC   41
#define GM    (GB * GT)  // 65536 rows
#define NGRP  2          // independent batch groups in the scan
#define GWG   4          // column-WGs per group (sync fan-in)
#define GRPB  16         // batches per group
#define CJS   64         // h-columns owned per scan workgroup

typedef __attribute__((ext_vector_type(16))) _Float16 v16h;
typedef __attribute__((ext_vector_type(8)))  float    v8f;
typedef __attribute__((ext_vector_type(4)))  int      v4i;

#ifndef USE_ASYNC_LDS
#define USE_ASYNC_LDS 1
#endif
#if USE_ASYNC_LDS && __has_builtin(__builtin_amdgcn_global_load_async_to_lds_b128) && \
    __has_builtin(__builtin_amdgcn_s_wait_asynccnt)
#define HAVE_ASYNC_LDS 1
// toolchain signature (probe-confirmed): (v4i AS1*, v4i AS3*, imm int, imm int)
typedef v4i __attribute__((address_space(1)))* async_gptr;
typedef v4i __attribute__((address_space(3)))* async_lptr;
#else
#define HAVE_ASYNC_LDS 0
#endif

// ---------------- fragment loaders (wave32 WMMA 16x16x32 f16) ----------------
// A 16x32 (MxK): lane m=lane&15; element s holds K = s + (lane>=16)*16
__device__ __forceinline__ v16h frag_a(const _Float16* __restrict__ p, int lda, int lane) {
  const _Float16* row = p + (size_t)(lane & 15) * lda + ((lane >> 4) << 4);
  return *(const v16h*)row;
}
// B 32x16 (KxN) with B[k,n] = W[n,k], W row-major [N,K]: lane n=lane&15 reads
// 16 contiguous K halves at k0 + 16*(lane>=16)  (32B aligned)
__device__ __forceinline__ v16h frag_b(const _Float16* __restrict__ p, int ldw, int lane) {
  const _Float16* row = p + (size_t)(lane & 15) * ldw + ((lane >> 4) << 4);
  return *(const v16h*)row;
}

__device__ __forceinline__ float sigm(float x) { return 1.0f / (1.0f + __expf(-x)); }

// ---------------- fp32 -> fp16 conversion ----------------
__global__ void k_cvt_f16(const float* __restrict__ s, _Float16* __restrict__ d, size_t n) {
  size_t stride = (size_t)gridDim.x * blockDim.x;
  for (size_t i = (size_t)blockIdx.x * blockDim.x + threadIdx.x; i < n; i += stride)
    d[i] = (_Float16)s[i];
}

// ---------------- generic WMMA GEMM: C[M,N] = A[M,K] * W[N,K]^T + bias -------
// 256 threads = 8 waves; one 16x16 output tile per wave.
template <bool HALF_OUT>
__global__ __launch_bounds__(256) void k_wmma_gemm(
    const _Float16* __restrict__ A, int lda,
    const _Float16* __restrict__ W, int ldw,
    const float* __restrict__ bias,
    void* __restrict__ Cout, int ldc, int M, int N, int K) {
  const int lane = threadIdx.x & 31;
  const int wave = threadIdx.x >> 5;
  const int ntiles = (N + 15) >> 4;
  const int tile = blockIdx.x * 8 + wave;
  const int mt = tile / ntiles, nt = tile % ntiles;
  if ((mt << 4) >= M) return;
  const int m0 = mt << 4, n0 = nt << 4;
  const _Float16* ap = A + (size_t)m0 * lda;
  const _Float16* wp = W + (size_t)n0 * ldw;
  v8f acc = {};
  for (int k0 = 0; k0 < K; k0 += 32) {
    __builtin_prefetch(ap + k0 + 64, 0, 0);   // global_prefetch_b8
    v16h a = frag_a(ap + k0, lda, lane);
    v16h b = frag_b(wp + k0, ldw, lane);
    acc = __builtin_amdgcn_wmma_f32_16x16x32_f16(false, a, false, b, (short)0, acc,
                                                 false, false);
  }
  const int n = n0 + (lane & 15);
  if (n < N) {
    const float bv = bias[n];
    const int mbase = m0 + ((lane >> 4) << 3);
    if (HALF_OUT) {
      _Float16* C = (_Float16*)Cout;
#pragma unroll
      for (int r = 0; r < 8; ++r)
        C[(size_t)(mbase + r) * ldc + n] = (_Float16)(acc[r] + bv);
    } else {
      float* C = (float*)Cout;
#pragma unroll
      for (int r = 0; r < 8; ++r)
        C[(size_t)(mbase + r) * ldc + n] = acc[r] + bv;
    }
  }
}

// ---------------- 4-WG barrier (atomic + hw cluster barrier) -----------------
__device__ __forceinline__ void group_sync(unsigned* bar, unsigned expected) {
  __syncthreads();
  if (threadIdx.x == 0) {
    __threadfence();                 // release (device scope)
    atomicAdd(bar, 1u);
    while (__hip_atomic_load(bar, __ATOMIC_RELAXED, __HIP_MEMORY_SCOPE_AGENT) < expected)
      __builtin_amdgcn_s_sleep(1);
    __threadfence();                 // acquire
  }
#if __has_builtin(__builtin_amdgcn_s_cluster_barrier)
  __builtin_amdgcn_s_cluster_barrier();  // s_barrier_signal/-wait -3 (NOP if unclustered)
#endif
  __syncthreads();
}

// ---------------- persistent GRU scan --------------------------------------
// 8 WGs = 2 batch groups (independent) x 4 column-WGs. WG (grp,cq) owns
// batches [grp*16,grp*16+16) and h columns [cq*64,cq*64+64) -> W_hh rows
// {j, 256+j, 512+j}: a 192x256 f16 slice (96KB) LDS-resident for all T.
// h state per group ping-pongs between two f16 [16,256] global slots (L2-hot).
__global__ __launch_bounds__(256) void k_gru_scan(
    const _Float16* __restrict__ gx,     // [B,T,768] f16 x-projection (+b_ih)
    const _Float16* __restrict__ Whh,    // [768,256] f16
    const float* __restrict__ bhh,       // [768]
    _Float16* __restrict__ hbuf,         // [NGRP][2][16][256] f16 ping-pong
    _Float16* __restrict__ hseq,         // [B,T,256] f16 output sequence
    unsigned* __restrict__ bars, int T) {
  __shared__ _Float16 Ws[3 * CJS][GH];   // 96 KB  (CDNA5 320KB LDS)
  __shared__ float gh[GRPB][3 * CJS];    // 12 KB
  const int tid = threadIdx.x, lane = tid & 31, wave = tid >> 5;
  const int grp = blockIdx.x >> 2;       // batch group
  const int cq  = blockIdx.x & 3;        // column quarter
  const int j0 = cq * CJS, b0 = grp * GRPB;
  unsigned* bar = bars + grp;
  _Float16* hb = hbuf + (size_t)grp * (2 * GRPB * GH);

  // Stage W_hh slice into LDS (async global->LDS path on gfx1250).
  for (int c = tid; c < 3 * CJS * (GH / 8); c += 256) {   // 16-byte chunks
    const int lr = c >> 5, ck = (c & 31) << 3;
    const int gr = (lr >> 6) * GH + j0 + (lr & 63);       // gate*256 + j
    const _Float16* src = Whh + (size_t)gr * GH + ck;
#if HAVE_ASYNC_LDS
    __builtin_amdgcn_global_load_async_to_lds_b128(
        (async_gptr)(uintptr_t)src,
        (async_lptr)(uint32_t)(uintptr_t)(void*)&Ws[lr][ck],
        0, 0);
#else
    *(float4*)&Ws[lr][ck] = *(const float4*)src;
#endif
  }
  float hprev[4] = {0.f, 0.f, 0.f, 0.f};                  // own (b,j) pairs
  for (int c = tid; c < GRPB * CJS; c += 256)             // zero slot 0 of h
    hb[(size_t)(c >> 6) * GH + j0 + (c & 63)] = (_Float16)0.0f;
#if HAVE_ASYNC_LDS
  __builtin_amdgcn_s_wait_asynccnt(0);
#endif
  unsigned epoch = 0;
  group_sync(bar, (++epoch) * GWG);      // h(-1)=0 visible in the group

  for (int t = 0; t < T; ++t) {
    const _Float16* hrd = hb + (size_t)(t & 1) * (GRPB * GH);
    _Float16* hwr = hb + (size_t)((t + 1) & 1) * (GRPB * GH);

    // gh[16][192] = h[16,256] x Ws^T : 12 N-tiles over 8 waves, 8 WMMAs each
    for (int nt = wave; nt < 12; nt += 8) {
      v8f acc = {};
#pragma unroll
      for (int k0 = 0; k0 < GH; k0 += 32) {
        v16h a = frag_a(hrd + k0, GH, lane);
        v16h b = *(const v16h*)&Ws[(nt << 4) + (lane & 15)][k0 + ((lane >> 4) << 4)];
        acc = __builtin_amdgcn_wmma_f32_16x16x32_f16(false, a, false, b, (short)0, acc,
                                                     false, false);
      }
      const int mb = (lane >> 4) << 3, nn = (nt << 4) + (lane & 15);
#pragma unroll
      for (int r = 0; r < 8; ++r) gh[mb + r][nn] = acc[r];
    }
    __syncthreads();

    // fused gate math: 16*64 (b,j) pairs / 256 threads = 4 per thread
#pragma unroll
    for (int i = 0; i < 4; ++i) {
      const int p = tid + (i << 8);
      const int b = p >> 6, c = p & 63, j = j0 + c, bg = b0 + b;
      const size_t gbase = ((size_t)bg * T + t) * G3;
      const float xr = (float)gx[gbase + j];
      const float xz = (float)gx[gbase + GH + j];
      const float xn = (float)gx[gbase + 2 * GH + j];
      const float hr = gh[b][c]           + bhh[j];
      const float hz = gh[b][CJS + c]     + bhh[GH + j];
      const float hn = gh[b][2 * CJS + c] + bhh[2 * GH + j];
      const float r = sigm(xr + hr);
      const float z = sigm(xz + hz);
      const float n = tanhf(xn + r * hn);
      const float hnew = (1.0f - z) * n + z * hprev[i];
      hprev[i] = hnew;
      const _Float16 hf = (_Float16)hnew;
      hwr[(size_t)b * GH + j] = hf;
      hseq[((size_t)bg * T + t) * GH + j] = hf;
    }
    group_sync(bar, (++epoch) * GWG);    // publish h(t) within the group
  }
}

// ---------------------------------------------------------------------------
extern "C" void kernel_launch(void* const* d_in, const int* in_sizes, int n_in,
                              void* d_out, int out_size, void* d_ws, size_t ws_size,
                              hipStream_t stream) {
  (void)in_sizes; (void)n_in; (void)out_size; (void)ws_size;
  const float* x    = (const float*)d_in[0];
  const float* Wih0 = (const float*)d_in[1];
  const float* Whh0 = (const float*)d_in[2];
  const float* bih0 = (const float*)d_in[3];
  const float* bhh0 = (const float*)d_in[4];
  const float* Wih1 = (const float*)d_in[5];
  const float* Whh1 = (const float*)d_in[6];
  const float* bih1 = (const float*)d_in[7];
  const float* bhh1 = (const float*)d_in[8];
  const float* Wfc  = (const float*)d_in[9];
  const float* bfc  = (const float*)d_in[10];
  float* out = (float*)d_out;

  char* w = (char*)d_ws;
  size_t off = 0;
  auto take = [&](size_t bytes) -> char* {
    char* p = w + off;
    off = (off + bytes + 255) & ~(size_t)255;
    return p;
  };
  unsigned*  bars  = (unsigned*)take(256);                       // 4 barrier counters
  _Float16*  xh    = (_Float16*)take((size_t)GM * GIN * 2);      //  67 MB
  _Float16*  gx    = (_Float16*)take((size_t)GM * G3 * 2);       // 100 MB (reused L0/L1)
  _Float16*  h1h   = (_Float16*)take((size_t)GM * GH * 2);       //  34 MB
  _Float16*  h2h   = (_Float16*)take((size_t)GM * GH * 2);       //  34 MB
  _Float16*  hbuf  = (_Float16*)take((size_t)NGRP * 2 * GRPB * GH * 2);
  _Float16*  Wih0h = (_Float16*)take((size_t)G3 * GIN * 2);
  _Float16*  Whh0h = (_Float16*)take((size_t)G3 * GH * 2);
  _Float16*  Wih1h = (_Float16*)take((size_t)G3 * GH * 2);
  _Float16*  Whh1h = (_Float16*)take((size_t)G3 * GH * 2);
  _Float16*  Wfch  = (_Float16*)take((size_t)48 * GH * 2);       // padded to 48 rows

  (void)hipMemsetAsync(bars, 0, 256, stream);

  auto cvt = [&](const float* s, _Float16* d, size_t n) {
    size_t b = (n + 1023) / 1024;
    if (b > 4096) b = 4096;
    k_cvt_f16<<<(unsigned)b, 256, 0, stream>>>(s, d, n);
  };
  cvt(x, xh, (size_t)GM * GIN);
  cvt(Wih0, Wih0h, (size_t)G3 * GIN);
  cvt(Whh0, Whh0h, (size_t)G3 * GH);
  cvt(Wih1, Wih1h, (size_t)G3 * GH);
  cvt(Whh1, Whh1h, (size_t)G3 * GH);
  cvt(Wfc,  Wfch,  (size_t)GNC * GH);     // rows 41..47 never stored from

  auto blocks = [](int M, int N) {
    int ntiles = (N + 15) >> 4;
    return ((M >> 4) * ntiles + 7) / 8;
  };

  // Layer 0
  k_wmma_gemm<true><<<blocks(GM, G3), 256, 0, stream>>>(xh, GIN, Wih0h, GIN, bih0,
                                                        gx, G3, GM, G3, GIN);
  k_gru_scan<<<NGRP * GWG, 256, 0, stream>>>(gx, Whh0h, bhh0, hbuf, h1h, bars + 0, GT);
  // Layer 1
  k_wmma_gemm<true><<<blocks(GM, G3), 256, 0, stream>>>(h1h, GH, Wih1h, GH, bih1,
                                                        gx, G3, GM, G3, GH);
  k_gru_scan<<<NGRP * GWG, 256, 0, stream>>>(gx, Whh1h, bhh1, hbuf, h2h, bars + 2, GT);
  // Classifier
  k_wmma_gemm<false><<<blocks(GM, GNC), 256, 0, stream>>>(h2h, GH, Wfch, GH, bfc,
                                                          out, GNC, GM, GNC, GH);
}
